// SelfAttention_w_RoPE_18743237280651
// MI455X (gfx1250) — matrix operand order repaired and checked by reading, not compiled, not run on hardware
//
#include <hip/hip_runtime.h>

// ---------------------------------------------------------------------------
// Types for CDNA5 WMMA (wave32)
// ---------------------------------------------------------------------------
typedef __attribute__((ext_vector_type(16))) __bf16 v16bf;
typedef __attribute__((ext_vector_type(8)))  float  v8f;

union FragBF {
    v16bf v;
    unsigned int u[8];
    unsigned short s[16];
};

static __device__ __forceinline__ unsigned short f2bf(float f) {
    union { float f; unsigned int u; } x;
    x.f = f;
    unsigned int u = x.u;
    u += 0x7fffu + ((u >> 16) & 1u);   // round-to-nearest-even
    return (unsigned short)(u >> 16);
}

static __device__ __forceinline__ v8f wmma_bf16(const FragBF& a, const FragBF& b, v8f c) {
    return __builtin_amdgcn_wmma_f32_16x16x32_bf16(
        /*neg_a=*/false, a.v, /*neg_b=*/false, b.v,
        /*c_mod=*/(short)0, c, /*reuse_a=*/false, /*reuse_b=*/false);
}

// ---------------------------------------------------------------------------
// CDNA5 async global->LDS staging (ASYNCcnt path), guarded by __has_builtin.
// Builtin signature (from hipcc diagnostic): (v4i AS1*, v4i AS3*, i32, i32).
// Falls back to b128 load + ds_store_b128 if the builtins are absent.
// ---------------------------------------------------------------------------
#if defined(__has_builtin)
#if __has_builtin(__builtin_amdgcn_global_load_async_to_lds_b128) && \
    __has_builtin(__builtin_amdgcn_s_wait_asynccnt)
#define HAVE_ASYNC_LDS 1
#endif
#endif
#ifndef HAVE_ASYNC_LDS
#define HAVE_ASYNC_LDS 0
#endif

typedef int async_v4i __attribute__((vector_size(16)));
typedef __attribute__((address_space(1))) async_v4i* async_gptr;
typedef __attribute__((address_space(3))) async_v4i* async_lptr;

static __device__ __forceinline__ void stage16(const unsigned short* g, unsigned short* l) {
#if HAVE_ASYNC_LDS
    __builtin_amdgcn_global_load_async_to_lds_b128(
        (async_gptr)g, (async_lptr)l, /*offset=*/0, /*cpol=*/0);
#else
    *reinterpret_cast<uint4*>(l) = *reinterpret_cast<const uint4*>(g);
#endif
}

static __device__ __forceinline__ void stage_fence() {
#if HAVE_ASYNC_LDS
    __builtin_amdgcn_s_wait_asynccnt(0);
#endif
}

// Problem constants
#define BB 8
#define TT 1024
#define CC 1024
#define HH 16
#define DD 64

// LDS pitches: multiples of 8 shorts (16B, async-B128 aligned) and chosen so
// 16 consecutive rows hit distinct banks (20 / 36 dword strides vs 64 banks).
#define PITCH32 40   // rows of 32 shorts (+8 pad) -> 80B
#define PITCH64 72   // rows of 64 shorts (+8 pad) -> 144B

// ---------------------------------------------------------------------------
// Elementwise fp32 -> bf16 (vectorized x4)
// ---------------------------------------------------------------------------
__global__ __launch_bounds__(256)
void conv_bf16(const float* __restrict__ src, unsigned short* __restrict__ dst)
{
    int i = (blockIdx.x * 256 + threadIdx.x) * 4;
    const float4 v = *reinterpret_cast<const float4*>(src + i);
    uint2 o;
    o.x = (unsigned int)f2bf(v.x) | ((unsigned int)f2bf(v.y) << 16);
    o.y = (unsigned int)f2bf(v.z) | ((unsigned int)f2bf(v.w) << 16);
    *reinterpret_cast<uint2*>(dst + i) = o;
}

// ---------------------------------------------------------------------------
// Transpose + convert: src fp32 [R][Cc] -> dst bf16 [Cc][R]   (32x32 tiles)
// ---------------------------------------------------------------------------
__global__ __launch_bounds__(256)
void convT_bf16(const float* __restrict__ src, unsigned short* __restrict__ dst,
                int R, int Cc)
{
    __shared__ float tile[32][33];
    const int bx = blockIdx.x * 32;   // column base in src
    const int by = blockIdx.y * 32;   // row base in src
    const int tx = threadIdx.x & 31;
    const int ty = threadIdx.x >> 5;  // 8 rows per pass
    for (int i = 0; i < 32; i += 8)
        tile[ty + i][tx] = src[(size_t)(by + ty + i) * Cc + bx + tx];
    __syncthreads();
    for (int i = 0; i < 32; i += 8)
        dst[(size_t)(bx + ty + i) * R + by + tx] = f2bf(tile[tx][ty + i]);
}

// ---------------------------------------------------------------------------
// GEMM: C(MxN fp32) = A(MxK bf16, row-major) * Bt(NxK bf16, i.e. B^T)
// 128x128 block tile, 256 threads (8 waves 2x4), K-step 32, async LDS staging.
// ---------------------------------------------------------------------------
__global__ __launch_bounds__(256)
void gemm_bf16(const unsigned short* __restrict__ A,
               const unsigned short* __restrict__ Bt,
               float* __restrict__ C, int M, int N, int K)
{
    __shared__ __align__(16) unsigned short ldsA[128][PITCH32];   // [m][k]
    __shared__ __align__(16) unsigned short ldsB[128][PITCH32];   // [n][k]

    const int tid  = threadIdx.x;
    const int wid  = tid >> 5;
    const int lane = tid & 31;
    const int hf   = lane >> 4;
    const int ln   = lane & 15;

    const int bm = blockIdx.y * 128;
    const int bn = blockIdx.x * 128;
    const int wm = (wid >> 2) * 64;
    const int wn = (wid & 3)  * 32;

    v8f acc[4][2];
    for (int i = 0; i < 4; ++i)
        for (int j = 0; j < 2; ++j)
            for (int e = 0; e < 8; ++e)
                acc[i][j][e] = 0.0f;

    for (int k0 = 0; k0 < K; k0 += 32) {
        // stage A & B tiles: 128 rows x 32 shorts = 512 x 16B chunks each
        for (int r = 0; r < 2; ++r) {
            int c2  = tid + r * 256;          // 0..511
            int row = c2 >> 2;                // 4 chunks per 64B row
            int kq  = (c2 & 3) << 3;          // short offset 0/8/16/24
            stage16(A  + (size_t)(bm + row) * K + k0 + kq, &ldsA[row][kq]);
            stage16(Bt + (size_t)(bn + row) * K + k0 + kq, &ldsB[row][kq]);
        }
        stage_fence();
        __syncthreads();

        // fragments (contiguous k order, k = 16*hf + s, same for A and B)
        FragBF afr[4];
        for (int mt = 0; mt < 4; ++mt) {
            int m = wm + mt * 16 + ln;
            for (int p = 0; p < 8; ++p) {
                int ks = 16 * hf + 2 * p;
                afr[mt].u[p] = *reinterpret_cast<const unsigned int*>(&ldsA[m][ks]);
            }
        }
        FragBF bfr[2];
        for (int nt = 0; nt < 2; ++nt) {
            int n = wn + nt * 16 + ln;
            for (int p = 0; p < 8; ++p)
                bfr[nt].u[p] = *reinterpret_cast<const unsigned int*>(&ldsB[n][16 * hf + 2 * p]);
        }
        for (int mt = 0; mt < 4; ++mt)
            for (int nt = 0; nt < 2; ++nt)
                acc[mt][nt] = wmma_bf16(afr[mt], bfr[nt], acc[mt][nt]);
        __syncthreads();
    }

    // epilogue: pointer-walk rows (no per-element 64-bit multiplies)
    for (int mt = 0; mt < 4; ++mt)
        for (int nt = 0; nt < 2; ++nt) {
            float* cp = C + (size_t)(bm + wm + mt * 16 + 8 * hf) * N
                          + (bn + wn + nt * 16 + ln);
            #pragma unroll
            for (int e = 0; e < 8; ++e) { *cp = acc[mt][nt][e]; cp += N; }
        }
}

// ---------------------------------------------------------------------------
// RoPE + repack: qkv fp32 [B,T,3C] -> Qb,Kb bf16 [B,H,T,D] (rotated),
//                                   Vt bf16 [B,H,D,T] (transposed).
// ---------------------------------------------------------------------------
__global__ __launch_bounds__(256)
void rope_pack(const float* __restrict__ qkv, const float* __restrict__ omega,
               unsigned short* __restrict__ Qb, unsigned short* __restrict__ Kb,
               unsigned short* __restrict__ Vt)
{
    int idx = blockIdx.x * 256 + threadIdx.x;
    int d2 = idx & 31;
    int h  = (idx >> 5) & 15;
    int t  = (idx >> 9) & 1023;
    int b  = idx >> 19;

    const size_t rowoff = (size_t)(b * TT + t) * (3 * CC);
    const int ch = h * DD + d2;
    float q1 = qkv[rowoff + ch];
    float q2 = qkv[rowoff + ch + 32];
    float k1 = qkv[rowoff + CC + ch];
    float k2 = qkv[rowoff + CC + ch + 32];
    float v1 = qkv[rowoff + 2 * CC + ch];
    float v2 = qkv[rowoff + 2 * CC + ch + 32];

    float ang = (float)t * omega[d2];
    float cs = cosf(ang);
    float sn = sinf(ang);

    size_t qoff = ((size_t)(b * HH + h) * TT + t) * DD + d2;     // [B,H,T,D]
    Qb[qoff]      = f2bf(q1 * cs - q2 * sn);
    Qb[qoff + 32] = f2bf(q2 * cs + q1 * sn);
    Kb[qoff]      = f2bf(k1 * cs - k2 * sn);
    Kb[qoff + 32] = f2bf(k2 * cs + k1 * sn);

    size_t voff = ((size_t)(b * HH + h) * DD + d2) * TT + t;     // [B,H,D,T]
    Vt[voff]           = f2bf(v1);
    Vt[voff + 32 * TT] = f2bf(v2);
}

// ---------------------------------------------------------------------------
// Flash attention: block = (b,h, 64 query rows), 4 waves x 16 rows.
// K/V tiles async-staged to LDS once per 32-key step (shared by all waves).
// Output written as bf16 (feeds final projection GEMM directly).
// ---------------------------------------------------------------------------
__global__ __launch_bounds__(128)
void attn(const unsigned short* __restrict__ Qb, const unsigned short* __restrict__ Kb,
          const unsigned short* __restrict__ Vt, unsigned short* __restrict__ Yb)
{
    const int blk  = blockIdx.x;
    const int qt   = blk & 15;
    const int bh   = blk >> 4;
    const int tid  = threadIdx.x;
    const int wid  = tid >> 5;
    const int lane = tid & 31;
    const int hf   = lane >> 4;
    const int ln   = lane & 15;
    const int q0   = qt * 64 + wid * 16;

    const unsigned short* Qp = Qb + (size_t)bh * TT * DD;
    const unsigned short* Kp = Kb + (size_t)bh * TT * DD;
    const unsigned short* Vp = Vt + (size_t)bh * DD * TT;

    __shared__ __align__(16) unsigned short ldsK[32][PITCH64];     // [key][d]
    __shared__ __align__(16) unsigned short ldsV[64][PITCH32];     // [d][key]
    __shared__ __align__(16) unsigned short pLDS[4][16][PITCH32];  // per-wave P

    // Q A-fragments (rows q0..q0+15, two D-chunks), kept in regs for all steps
    FragBF aq[2];
    for (int c = 0; c < 2; ++c)
        for (int p = 0; p < 8; ++p) {
            int ks = 32 * c + 16 * hf + 2 * p;
            aq[c].u[p] = *reinterpret_cast<const unsigned int*>(
                Qp + (size_t)(q0 + ln) * DD + ks);
        }

    v8f oacc[4];
    float mrun[8], lrun[8];
    for (int nt = 0; nt < 4; ++nt)
        for (int e = 0; e < 8; ++e) oacc[nt][e] = 0.0f;
    for (int e = 0; e < 8; ++e) { mrun[e] = -3.0e38f; lrun[e] = 0.0f; }

    const float scale = 0.125f;   // 1/sqrt(64)

    for (int kt = 0; kt < TT / 32; ++kt) {
        const int kbase = kt * 32;

        // stage K tile (32x64 bf16 = 256 x 16B) and V^T tile (64x32 = 256 x 16B)
        for (int r = 0; r < 2; ++r) {
            int c2 = tid + r * 128;                // 0..255
            int kr = c2 >> 3, ko = (c2 & 7) << 3;  // K: 8 chunks per 128B row
            stage16(Kp + (size_t)(kbase + kr) * DD + ko, &ldsK[kr][ko]);
            int dv = c2 >> 2, kv = (c2 & 3) << 3;  // V: 4 chunks per 64B row
            stage16(Vp + (size_t)dv * TT + kbase + kv, &ldsV[dv][kv]);
        }
        stage_fence();
        __syncthreads();

        // S = Q * K^T for 32 keys (two 16-key n-tiles, two D-chunks)
        v8f s0, s1;
        for (int e = 0; e < 8; ++e) { s0[e] = 0.0f; s1[e] = 0.0f; }
        for (int c = 0; c < 2; ++c) {
            FragBF bk0, bk1;
            for (int p = 0; p < 8; ++p) {
                int dof = 32 * c + 16 * hf + 2 * p;
                bk0.u[p] = *reinterpret_cast<const unsigned int*>(&ldsK[ln][dof]);
                bk1.u[p] = *reinterpret_cast<const unsigned int*>(&ldsK[16 + ln][dof]);
            }
            s0 = wmma_bf16(aq[c], bk0, s0);
            s1 = wmma_bf16(aq[c], bk1, s1);
        }

        // online softmax (row = element; reduce across 16 lanes of each half)
        float corr[8];
        for (int e = 0; e < 8; ++e) {
            float v0 = s0[e] * scale;
            float v1 = s1[e] * scale;
            float mx = fmaxf(v0, v1);
            for (int m = 1; m < 16; m <<= 1)
                mx = fmaxf(mx, __shfl_xor(mx, m, 32));
            float mnew = fmaxf(mrun[e], mx);
            corr[e] = __expf(mrun[e] - mnew);
            float p0 = __expf(v0 - mnew);
            float p1 = __expf(v1 - mnew);
            s0[e] = p0;
            s1[e] = p1;
            float rsum = p0 + p1;
            for (int m = 1; m < 16; m <<= 1)
                rsum += __shfl_xor(rsum, m, 32);
            lrun[e] = lrun[e] * corr[e] + rsum;
            mrun[e] = mnew;
        }
        for (int nt = 0; nt < 4; ++nt)
            for (int e = 0; e < 8; ++e) oacc[nt][e] *= corr[e];

        // P: C-layout -> LDS (per-wave region; same-wave DS ops stay in order)
        for (int e = 0; e < 8; ++e) {
            int row = e + 8 * hf;
            pLDS[wid][row][ln]      = f2bf(s0[e]);
            pLDS[wid][row][16 + ln] = f2bf(s1[e]);
        }
        FragBF pf;
        for (int p = 0; p < 8; ++p) {
            int ks = 16 * hf + 2 * p;
            pf.u[p] = *reinterpret_cast<const unsigned int*>(&pLDS[wid][ln][ks]);
        }

        // O += P * V   (V^T in LDS -> contiguous B-fragment loads)
        for (int nt = 0; nt < 4; ++nt) {
            FragBF vf;
            for (int p = 0; p < 8; ++p)
                vf.u[p] = *reinterpret_cast<const unsigned int*>(
                    &ldsV[nt * 16 + ln][16 * hf + 2 * p]);
            oacc[nt] = wmma_bf16(pf, vf, oacc[nt]);
        }
        __syncthreads();   // protect ldsK/ldsV before next stage
    }

    // epilogue: Yb[b, t, h*64 + d] bf16
    const int b  = bh >> 4;
    const int hh = bh & 15;
    for (int e = 0; e < 8; ++e) {
        float inv = 1.0f / lrun[e];
        int row = q0 + e + 8 * hf;
        unsigned short* yrow = Yb + ((size_t)(b * TT + row)) * CC + hh * DD;
        for (int nt = 0; nt < 4; ++nt)
            yrow[nt * 16 + ln] = f2bf(oacc[nt][e] * inv);
    }
}

// ---------------------------------------------------------------------------
// Launch
// ---------------------------------------------------------------------------
extern "C" void kernel_launch(void* const* d_in, const int* in_sizes, int n_in,
                              void* d_out, int out_size, void* d_ws, size_t ws_size,
                              hipStream_t stream)
{
    (void)in_sizes; (void)n_in; (void)out_size; (void)ws_size;
    const float* x      = (const float*)d_in[0];
    const float* W_attn = (const float*)d_in[1];
    const float* W_proj = (const float*)d_in[2];
    const float* omega  = (const float*)d_in[3];
    float* out = (float*)d_out;

    char* ws = (char*)d_ws;
    size_t off = 0;
    float* qkv = (float*)(ws + off);                  off += (size_t)BB * TT * 3 * CC * 4;  // 96 MB
    unsigned short* xb  = (unsigned short*)(ws + off); off += (size_t)BB * TT * CC * 2;     // 16 MB
    unsigned short* Wat = (unsigned short*)(ws + off); off += (size_t)CC * 3 * CC * 2;      //  6 MB
    unsigned short* Wpt = (unsigned short*)(ws + off); off += (size_t)CC * CC * 2;          //  2 MB
    unsigned short* Qb  = (unsigned short*)(ws + off); off += (size_t)BB * HH * TT * DD * 2;// 16 MB
    unsigned short* Kb  = (unsigned short*)(ws + off); off += (size_t)BB * HH * TT * DD * 2;// 16 MB
    unsigned short* Vt  = (unsigned short*)(ws + off); off += (size_t)BB * HH * DD * TT * 2;// 16 MB
    unsigned short* Yb  = (unsigned short*)(ws + off); off += (size_t)BB * TT * CC * 2;     // 16 MB

    // 0) precision pre-passes: x -> bf16, weights -> transposed bf16
    conv_bf16<<<(BB * TT * CC) / (256 * 4), 256, 0, stream>>>(x, xb);
    {
        dim3 g(3 * CC / 32, CC / 32);
        convT_bf16<<<g, 256, 0, stream>>>(W_attn, Wat, CC, 3 * CC);
    }
    {
        dim3 g(CC / 32, CC / 32);
        convT_bf16<<<g, 256, 0, stream>>>(W_proj, Wpt, CC, CC);
    }
    // 1) QKV projection: (8192x1024) @ (1024x3072)
    {
        dim3 grid(3 * CC / 128, BB * TT / 128);
        gemm_bf16<<<grid, 256, 0, stream>>>(xb, Wat, qkv, BB * TT, 3 * CC, CC);
    }
    // 2) RoPE + pack
    rope_pack<<<(BB * TT * HH * 32) / 256, 256, 0, stream>>>(qkv, omega, Qb, Kb, Vt);
    // 3) Flash attention
    attn<<<BB * HH * (TT / 64), 128, 0, stream>>>(Qb, Kb, Vt, Yb);
    // 4) Output projection: (8192x1024) @ (1024x1024)
    {
        dim3 grid(CC / 128, BB * TT / 128);
        gemm_bf16<<<grid, 256, 0, stream>>>(Yb, Wpt, out, BB * TT, CC, CC);
    }
}
